// PINN_Burgers2D_89146341195842
// MI455X (gfx1250) — matrix operand order repaired and checked by reading, not compiled, not run on hardware
//
#include <hip/hip_runtime.h>
#include <hip/hip_bf16.h>

// Fully-fused MLP (PINN Burgers2D) for gfx1250 / MI455X.
// 4 -> 128 -> (128x128)x5 -> 2, swish, f16 WMMA w/ f32 accumulate.
// Block = 512 threads = 16 wave32; each block processes 256 points; each wave
// owns a 16-row M-slice. Activations live in LDS for the whole network;
// weights are double-buffered in LDS with register-prefetch of the next layer
// overlapping the current layer's WMMAs. Bias is folded into the WMMA C init.

typedef __attribute__((ext_vector_type(16))) _Float16 v16h;
typedef __attribute__((ext_vector_type(8)))  _Float16 v8h;
typedef __attribute__((ext_vector_type(8)))  float    v8f;

#define MLP_WIDTH    128
#define ROWS_PER_BLK 256
#define THREADS      512
#define STRIDE       136   // padded f16 row stride: 272 B = 68 dwords -> 4-bank shift/row
#define WSLOT        (MLP_WIDTH * STRIDE)

__device__ __forceinline__ v16h load_frag_A(const _Float16* p) {
    // A 16x32 f16: lanes 0-15 hold K {0..7,16..23}, lanes 16-31 {8..15,24..31}
    union { v16h v; v8h h[2]; } u;
    u.h[0] = *(const v8h*)(p);        // ds_load_b128
    u.h[1] = *(const v8h*)(p + 16);
    return u.v;
}

__device__ __forceinline__ v16h load_frag_B(const _Float16* p) {
    // B 32x16 f16: lane n, lanes 0-15 hold K {0..7,16..23}, lanes 16-31 {8..15,24..31}
    union { v16h v; v8h h[2]; } u;
    u.h[0] = *(const v8h*)(p);
    u.h[1] = *(const v8h*)(p + 16);
    return u.v;
}

__device__ __forceinline__ v8f splat8(float x) {
    v8f v;
    #pragma unroll
    for (int i = 0; i < 8; ++i) v[i] = x;
    return v;
}

__device__ __forceinline__ float swish(float x) {
    return x / (1.0f + __expf(-x));   // x * sigmoid(x) via v_exp_f32
}

// One dense layer: sAct[256 x K] (K = 32*kChunks) @ sWT^T + bias -> swish -> sAct in place
__device__ __forceinline__ void dense_swish_128(_Float16* sAct, const _Float16* sWT,
                                                const float* __restrict__ bias,
                                                int wv, int lane, int kChunks) {
    const int m  = lane & 15;
    const int hi = lane >> 4;
    const _Float16* aRow = sAct + (wv * 16 + m) * STRIDE;

    v8f acc[8];
    #pragma unroll
    for (int nt = 0; nt < 8; ++nt) acc[nt] = splat8(bias[nt * 16 + m]);  // bias folded into C

    for (int kc = 0; kc < kChunks; ++kc) {
        const v16h a = load_frag_A(aRow + kc * 32 + hi * 8);
        #pragma unroll
        for (int nt = 0; nt < 8; ++nt) {
            const v16h b = load_frag_B(sWT + (nt * 16 + m) * STRIDE + kc * 32 + hi * 8);
            acc[nt] = __builtin_amdgcn_wmma_f32_16x16x32_f16(
                false, a, false, b, (short)0, acc[nt], false, false);
        }
    }

    // D layout: lane&15 = N within tile, hi*8+j = M
    #pragma unroll
    for (int nt = 0; nt < 8; ++nt) {
        _Float16* wr = sAct + (wv * 16 + hi * 8) * STRIDE + nt * 16 + m;
        #pragma unroll
        for (int j = 0; j < 8; ++j) {
            wr[j * STRIDE] = (_Float16)swish(acc[nt][j]);
        }
    }
}

__global__ __launch_bounds__(THREADS)
void PINN_Burgers2D_fused_mlp_wmma(const float* __restrict__ X,
                                   const float* __restrict__ nu_min_p,
                                   const float* __restrict__ nu_max_p,
                                   const float* __restrict__ W0,   // [4][128]
                                   const float* __restrict__ b0,   // [128]
                                   const float* __restrict__ Wh,   // [5][128][128]
                                   const float* __restrict__ bh,   // [5][128]
                                   const float* __restrict__ WL,   // [128][2]
                                   const float* __restrict__ bL,   // [2]
                                   float* __restrict__ out,        // [N][2]
                                   int nPoints) {
    extern __shared__ __align__(16) char smem_raw[];
    _Float16* sAct = (_Float16*)smem_raw;            // 256 x STRIDE f16
    _Float16* sW0  = sAct + ROWS_PER_BLK * STRIDE;   // weight buffer 0 (W^T[n][k])
    _Float16* sW1  = sW0 + WSLOT;                    // weight buffer 1

    const int tid  = threadIdx.x;
    const int lane = tid & 31;
    const int wv   = tid >> 5;            // wave id 0..15
    const int m    = lane & 15;
    const int hi   = lane >> 4;
    const int row0 = blockIdx.x * ROWS_PER_BLK;

    // ---- stage layer-0 activations: rescaled inputs, K padded 4 -> 32 ----
    const float nu_min = nu_min_p[0];
    const float nu_max = nu_max_p[0];
    const float nu_scl = 2.0f / (nu_max - nu_min);
    #pragma unroll
    for (int i = 0; i < (ROWS_PER_BLK * 32) / THREADS; ++i) {
        const int idx = tid + i * THREADS;
        const int r = idx >> 5, k = idx & 31;
        const int rg = row0 + r;
        float v = 0.0f;
        if (k < 4 && rg < nPoints) {
            const float x = X[rg * 4 + k];
            if (k <= 1)      v = x;                       // x,y already in [-1,1]
            else if (k == 2) v = 2.0f * x - 1.0f;         // t in [0,1]
            else             v = nu_scl * (x - nu_min) - 1.0f;
        }
        sAct[r * STRIDE + k] = (_Float16)v;
    }
    // ---- stage W0^T padded into sW0[n][k], k<4 valid, rest zero ----
    #pragma unroll
    for (int i = 0; i < (MLP_WIDTH * 32) / THREADS; ++i) {
        const int idx = tid + i * THREADS;
        const int n = idx >> 5, k = idx & 31;
        const float v = (k < 4) ? W0[k * MLP_WIDTH + n] : 0.0f;
        sW0[n * STRIDE + k] = (_Float16)v;
    }
    __syncthreads();

    float pf[32];

    // ---- layer 0 (K=32), prefetch Wh[0] during compute, store to sW1 ----
    #pragma unroll
    for (int i = 0; i < 32; ++i) pf[i] = Wh[tid + i * THREADS];   // idx = k*128+n
    asm volatile("" ::: "memory");   // keep prefetch issued before the WMMA stream
    dense_swish_128(sAct, sW0, b0, wv, lane, 1);
    #pragma unroll
    for (int i = 0; i < 32; ++i) {
        const int idx = tid + i * THREADS;
        sW1[(idx & 127) * STRIDE + (idx >> 7)] = (_Float16)pf[i];  // transpose: W^T[n][k]
    }
    __syncthreads();

    // ---- 5 hidden layers (K=128), double-buffered weights ----
    for (int L = 0; L < 5; ++L) {
        _Float16* bufR = (L & 1) ? sW0 : sW1;   // L=0 reads sW1 (staged above)
        _Float16* bufW = (L & 1) ? sW1 : sW0;
        if (L < 4) {
            #pragma unroll
            for (int i = 0; i < 32; ++i)
                pf[i] = Wh[(L + 1) * (MLP_WIDTH * MLP_WIDTH) + tid + i * THREADS];
        } else {
            // prefetch output weights WL^T padded to 16 x 128
            #pragma unroll
            for (int i = 0; i < 4; ++i) {
                const int idx = tid + i * THREADS;
                const int n = idx >> 7, k = idx & 127;
                pf[i] = (n < 2) ? WL[k * 2 + n] : 0.0f;
            }
        }
        asm volatile("" ::: "memory");
        dense_swish_128(sAct, bufR, bh + L * MLP_WIDTH, wv, lane, 4);
        if (L < 4) {
            #pragma unroll
            for (int i = 0; i < 32; ++i) {
                const int idx = tid + i * THREADS;
                bufW[(idx & 127) * STRIDE + (idx >> 7)] = (_Float16)pf[i];
            }
        } else {
            #pragma unroll
            for (int i = 0; i < 4; ++i) {
                const int idx = tid + i * THREADS;
                bufW[(idx >> 7) * STRIDE + (idx & 127)] = (_Float16)pf[i];
            }
        }
        __syncthreads();
    }

    // ---- output layer: 128 -> 2 (N padded to 16), weights in sW0 ----
    {
        const _Float16* aRow = sAct + (wv * 16 + m) * STRIDE;
        const float bv = (m < 2) ? bL[m] : 0.0f;
        v8f acc = splat8(bv);
        #pragma unroll
        for (int kc = 0; kc < 4; ++kc) {
            const v16h a = load_frag_A(aRow + kc * 32 + hi * 8);
            const v16h b = load_frag_B(sW0 + m * STRIDE + kc * 32 + hi * 8);
            acc = __builtin_amdgcn_wmma_f32_16x16x32_f16(
                false, a, false, b, (short)0, acc, false, false);
        }
        if (m < 2) {
            #pragma unroll
            for (int j = 0; j < 8; ++j) {
                const int rg = row0 + wv * 16 + hi * 8 + j;
                if (rg < nPoints) out[rg * 2 + m] = acc[j];
            }
        }
    }
}

extern "C" void kernel_launch(void* const* d_in, const int* in_sizes, int n_in,
                              void* d_out, int out_size, void* d_ws, size_t ws_size,
                              hipStream_t stream) {
    const float* X      = (const float*)d_in[0];
    const float* nu_min = (const float*)d_in[1];
    const float* nu_max = (const float*)d_in[2];
    const float* W0     = (const float*)d_in[3];
    const float* b0     = (const float*)d_in[4];
    const float* Wh     = (const float*)d_in[5];
    const float* bh     = (const float*)d_in[6];
    const float* WL     = (const float*)d_in[7];
    const float* bL     = (const float*)d_in[8];
    float* out = (float*)d_out;

    const int nPoints = in_sizes[0] / 4;
    const int nBlocks = (nPoints + ROWS_PER_BLK - 1) / ROWS_PER_BLK;
    // acts (256*136) + 2 weight buffers (2*128*136), f16
    const size_t shmem = (size_t)(ROWS_PER_BLK * STRIDE + 2 * WSLOT) * sizeof(_Float16); // 136 KB

    PINN_Burgers2D_fused_mlp_wmma<<<nBlocks, THREADS, shmem, stream>>>(
        X, nu_min, nu_max, W0, b0, Wh, bh, WL, bL, out, nPoints);
}
